// GIN_36696200577384
// MI455X (gfx1250) — matrix operand order repaired and checked by reading, not compiled, not run on hardware
//
#include <hip/hip_runtime.h>
#include <hip/hip_bf16.h>

#define N_NODES 50000
#define N_EDGES 800000
#define F_DIM   128
#define H_DIM   256
#define C_DIM   40
#define C_PAD   48
#define BN_EPS  1e-5f

typedef __attribute__((ext_vector_type(16))) __bf16 v16bf;
typedef __attribute__((ext_vector_type(8)))  float  v8f;

union alignas(32) BF16Frag {
    unsigned short u[16];
    uint4          q[2];
    v16bf          v;
};

__device__ __forceinline__ unsigned short f2bf(float f) {
    // round-to-nearest-even f32 -> bf16
    unsigned int u = __float_as_uint(f);
    u += 0x7fffu + ((u >> 16) & 1u);
    return (unsigned short)(u >> 16);
}

// ---------------------------------------------------------------- utilities
__global__ void zero_kernel(float* p, long n) {
    long i = (long)blockIdx.x * blockDim.x + threadIdx.x;
    long s = (long)gridDim.x * blockDim.x;
    for (; i < n; i += s) p[i] = 0.0f;
}

// W is [K][Hreal] row-major f32; Wt is [Hpad][K] bf16 (transposed, zero-padded)
__global__ void pack_w_kernel(const float* __restrict__ W,
                              unsigned short* __restrict__ Wt,
                              int K, int Hreal, int Hpad) {
    int i = blockIdx.x * blockDim.x + threadIdx.x;
    if (i >= K * Hpad) return;
    int h = i / K;
    int k = i - h * K;
    float v = (h < Hreal) ? W[(long)k * Hreal + h] : 0.0f;
    Wt[(long)h * K + k] = f2bf(v);
}

// ------------------------------------------------- edge scatter (segment_sum)
// one wave per edge: 32 lanes x float4 covers 128 floats per pass
__global__ __launch_bounds__(256)
void scatter_add_kernel(const float* __restrict__ X,
                        const int* __restrict__ src,
                        const int* __restrict__ dst,
                        float* __restrict__ agg, int E, int F) {
    int wave = (blockIdx.x * blockDim.x + threadIdx.x) >> 5;
    int lane = threadIdx.x & 31;
    if (wave >= E) return;
    int s = src[wave];
    int d = dst[wave];
    const float* xs = X + (long)s * F;
    float*       ad = agg + (long)d * F;
    for (int f = lane * 4; f < F; f += 128) {
        float4 v = *(const float4*)(xs + f);
        atomicAdd(ad + f + 0, v.x);
        atomicAdd(ad + f + 1, v.y);
        atomicAdd(ad + f + 2, v.z);
        atomicAdd(ad + f + 3, v.w);
    }
}

// ---------------------------------------------------------- fused WMMA GEMM
// mode 0: a[m][k] = (1+eps)*A1[m][k] + A2[m][k]          (GIN combine)
// mode 1: a[m][k] = relu(A1[m][k]*bscale[k] + bshift[k]) (BN+ReLU on input)
// Y = a @ Wt^T + bias  (Wt is [HoutPad][K] bf16); optional output ReLU;
// optional per-column sum/sumsq accumulation for the next BatchNorm.
__global__ __launch_bounds__(256)
void gemm_fused_kernel(const float* __restrict__ A1,
                       const float* __restrict__ A2,
                       const float* __restrict__ bscale,
                       const float* __restrict__ bshift,
                       const float* __restrict__ epsPtr,
                       const unsigned short* __restrict__ Wt,
                       const float* __restrict__ bias,
                       float* __restrict__ Y,
                       float* __restrict__ ssum,
                       float* __restrict__ ssq,
                       int K, int kLog, int Hout, int HoutPad,
                       int mode, int reluOut) {
    __shared__ unsigned short ldsA[16 * (H_DIM + 8)];  // 16 rows, padded stride
    const int Kpad = K + 8;
    const int tid  = threadIdx.x;
    const long rowbase = (long)blockIdx.x * 16;

    float alpha = (mode == 0) ? (1.0f + epsPtr[0]) : 0.0f;

    // cooperative load + input transform + f32->bf16 into LDS
    const int total = 16 << kLog;
    const int kmask = (1 << kLog) - 1;
    for (int idx = tid; idx < total; idx += blockDim.x) {
        int m = idx >> kLog;
        int k = idx & kmask;
        long row = rowbase + m;
        float v;
        if (mode == 0) {
            v = alpha * A1[row * K + k] + A2[row * K + k];
        } else {
            float t = A1[row * K + k] * bscale[k] + bshift[k];
            v = t > 0.0f ? t : 0.0f;
        }
        ldsA[m * Kpad + k] = f2bf(v);
    }
    __syncthreads();

    const int lane   = tid & 31;
    const int wv     = tid >> 5;
    const int laneHi = lane >> 4;      // 0 or 1
    const int n      = lane & 15;
    const int numCT  = HoutPad >> 4;
    const int nkt    = K >> 5;

    // A fragment base: row = n; K-chunks {kb..kb+7, kb+16..kb+23}, kb=laneHi*8
    const unsigned short* arow = ldsA + n * Kpad + laneHi * 8;

    for (int ct = wv; ct < numCT; ct += 8) {
        int h0 = ct << 4;
        v8f c = {0.f, 0.f, 0.f, 0.f, 0.f, 0.f, 0.f, 0.f};
        // B fragment base: col = n; K-chunks {kb..kb+7, kb+16..kb+23}, kb=laneHi*8
        const unsigned short* bcol = Wt + (long)(h0 + n) * K + laneHi * 8;
        for (int kt = 0; kt < nkt; ++kt) {
            BF16Frag a, b;
            a.q[0] = *(const uint4*)(arow + kt * 32);
            a.q[1] = *(const uint4*)(arow + kt * 32 + 16);
            b.q[0] = *(const uint4*)(bcol + kt * 32);
            b.q[1] = *(const uint4*)(bcol + kt * 32 + 16);
            c = __builtin_amdgcn_wmma_f32_16x16x32_bf16(
                    false, a.v, false, b.v, (short)0, c, false, false);
        }
        int col = h0 + n;
        if (col < Hout) {
            float bv = bias[col];
            float s = 0.0f, q = 0.0f;
#pragma unroll
            for (int r = 0; r < 8; ++r) {
                long row = rowbase + laneHi * 8 + r;   // C/D VGPR layout
                float v = c[r] + bv;
                if (reluOut) v = v > 0.0f ? v : 0.0f;
                Y[row * Hout + col] = v;
                s += v;
                q += v * v;
            }
            if (ssum) {   // per-column BN statistics (8 rows reduced locally)
                atomicAdd(&ssum[col], s);
                atomicAdd(&ssq[col], q);
            }
        }
    }
}

// --------------------------------------------------------- BN finalize
__global__ void bn_finalize_kernel(const float* __restrict__ ssum,
                                   const float* __restrict__ ssq,
                                   const float* __restrict__ g,
                                   const float* __restrict__ be,
                                   float* __restrict__ scale,
                                   float* __restrict__ shift,
                                   int H, float invN) {
    int i = blockIdx.x * blockDim.x + threadIdx.x;
    if (i >= H) return;
    float mean = ssum[i] * invN;
    float var  = ssq[i] * invN - mean * mean;
    float s    = g[i] * rsqrtf(var + BN_EPS);
    scale[i] = s;
    shift[i] = be[i] - mean * s;
}

// ---------------------------------------------------------------- launch
extern "C" void kernel_launch(void* const* d_in, const int* in_sizes, int n_in,
                              void* d_out, int out_size, void* d_ws, size_t ws_size,
                              hipStream_t stream) {
    const float* x    = (const float*)d_in[0];
    const int*   ei   = (const int*)d_in[1];
    const float* eps1 = (const float*)d_in[2];
    const float* W1   = (const float*)d_in[3];
    const float* b1   = (const float*)d_in[4];
    const float* g1   = (const float*)d_in[5];
    const float* be1  = (const float*)d_in[6];
    const float* W2   = (const float*)d_in[7];
    const float* b2   = (const float*)d_in[8];
    const float* eps2 = (const float*)d_in[9];
    const float* W3   = (const float*)d_in[10];
    const float* b3   = (const float*)d_in[11];
    const float* g2   = (const float*)d_in[12];
    const float* be2  = (const float*)d_in[13];
    const float* W4   = (const float*)d_in[14];
    const float* b4   = (const float*)d_in[15];
    const float* g3   = (const float*)d_in[16];
    const float* be3  = (const float*)d_in[17];
    const float* Wf   = (const float*)d_in[18];
    const float* bfp  = (const float*)d_in[19];
    const int* srcI = ei;            // edge_index[0]
    const int* dstI = ei + N_EDGES;  // edge_index[1]

    // workspace layout (floats)
    float* ws = (float*)d_ws;
    size_t off = 0;
    float* agg1 = ws + off; off += (size_t)N_NODES * F_DIM;  // layer-1 aggregate
    float* bufA = ws + off; off += (size_t)N_NODES * H_DIM;  // y1, then y3
    float* bufB = ws + off; off += (size_t)N_NODES * H_DIM;  // x2 (post conv1+relu)
    float* bufC = ws + off; off += (size_t)N_NODES * H_DIM;  // agg2, then y4
    float* stats = ws + off; off += 6 * H_DIM;               // 3x (sum, sumsq)
    float* bnp   = ws + off; off += 6 * H_DIM;               // 3x (scale, shift)
    unsigned short* w1t = (unsigned short*)(ws + off);
    unsigned short* w2t = w1t + (size_t)H_DIM * F_DIM;
    unsigned short* w3t = w2t + (size_t)H_DIM * H_DIM;
    unsigned short* w4t = w3t + (size_t)H_DIM * H_DIM;
    unsigned short* wft = w4t + (size_t)H_DIM * H_DIM;       // [C_PAD][H_DIM]

    const dim3 B256(256);
    const int gemmGrid = N_NODES / 16;              // 3125, exact
    const int edgeGrid = (N_EDGES + 7) / 8;         // one wave per edge

    // pack weights (f32 [K][H] -> bf16 [Hpad][K])
    pack_w_kernel<<<(F_DIM * H_DIM + 255) / 256, B256, 0, stream>>>(W1, w1t, F_DIM, H_DIM, H_DIM);
    pack_w_kernel<<<(H_DIM * H_DIM + 255) / 256, B256, 0, stream>>>(W2, w2t, H_DIM, H_DIM, H_DIM);
    pack_w_kernel<<<(H_DIM * H_DIM + 255) / 256, B256, 0, stream>>>(W3, w3t, H_DIM, H_DIM, H_DIM);
    pack_w_kernel<<<(H_DIM * H_DIM + 255) / 256, B256, 0, stream>>>(W4, w4t, H_DIM, H_DIM, H_DIM);
    pack_w_kernel<<<(H_DIM * C_PAD + 255) / 256, B256, 0, stream>>>(Wf, wft, H_DIM, C_DIM, C_PAD);

    // zero accumulators (must be re-zeroed every call: atomics accumulate)
    zero_kernel<<<2048, B256, 0, stream>>>(agg1, (long)N_NODES * F_DIM);
    zero_kernel<<<8, B256, 0, stream>>>(stats, 6 * H_DIM);

    // ---- GIN conv 1 ----
    scatter_add_kernel<<<edgeGrid, B256, 0, stream>>>(x, srcI, dstI, agg1, N_EDGES, F_DIM);
    // y1 = ((1+eps1)x + agg1) @ W1 + b1, accumulate BN1 stats
    gemm_fused_kernel<<<gemmGrid, B256, 0, stream>>>(
        x, agg1, nullptr, nullptr, eps1, w1t, b1, bufA,
        stats + 0 * H_DIM, stats + 1 * H_DIM,
        F_DIM, 7, H_DIM, H_DIM, /*mode=*/0, /*reluOut=*/0);
    bn_finalize_kernel<<<1, B256, 0, stream>>>(
        stats + 0 * H_DIM, stats + 1 * H_DIM, g1, be1,
        bnp + 0 * H_DIM, bnp + 1 * H_DIM, H_DIM, 1.0f / N_NODES);
    // x2 = relu( relu(bn1(y1)) @ W2 + b2 )   (inter-layer relu fused as reluOut)
    gemm_fused_kernel<<<gemmGrid, B256, 0, stream>>>(
        bufA, nullptr, bnp + 0 * H_DIM, bnp + 1 * H_DIM, nullptr, w2t, b2, bufB,
        nullptr, nullptr, H_DIM, 8, H_DIM, H_DIM, /*mode=*/1, /*reluOut=*/1);

    // ---- GIN conv 2 ----
    zero_kernel<<<2048, B256, 0, stream>>>(bufC, (long)N_NODES * H_DIM);
    scatter_add_kernel<<<edgeGrid, B256, 0, stream>>>(bufB, srcI, dstI, bufC, N_EDGES, H_DIM);
    // y3 = ((1+eps2)x2 + agg2) @ W3 + b3, accumulate BN2 stats
    gemm_fused_kernel<<<gemmGrid, B256, 0, stream>>>(
        bufB, bufC, nullptr, nullptr, eps2, w3t, b3, bufA,
        stats + 2 * H_DIM, stats + 3 * H_DIM,
        H_DIM, 8, H_DIM, H_DIM, /*mode=*/0, /*reluOut=*/0);
    bn_finalize_kernel<<<1, B256, 0, stream>>>(
        stats + 2 * H_DIM, stats + 3 * H_DIM, g2, be2,
        bnp + 2 * H_DIM, bnp + 3 * H_DIM, H_DIM, 1.0f / N_NODES);
    // y4 = relu(bn2(y3)) @ W4 + b4, accumulate BN3 stats
    gemm_fused_kernel<<<gemmGrid, B256, 0, stream>>>(
        bufA, nullptr, bnp + 2 * H_DIM, bnp + 3 * H_DIM, nullptr, w4t, b4, bufC,
        stats + 4 * H_DIM, stats + 5 * H_DIM,
        H_DIM, 8, H_DIM, H_DIM, /*mode=*/1, /*reluOut=*/0);
    bn_finalize_kernel<<<1, B256, 0, stream>>>(
        stats + 4 * H_DIM, stats + 5 * H_DIM, g3, be3,
        bnp + 4 * H_DIM, bnp + 5 * H_DIM, H_DIM, 1.0f / N_NODES);

    // ---- classifier: out = relu(bn3(y4)) @ Wf + bf  -> [N, 40] ----
    gemm_fused_kernel<<<gemmGrid, B256, 0, stream>>>(
        bufC, nullptr, bnp + 4 * H_DIM, bnp + 5 * H_DIM, nullptr, wft, bfp,
        (float*)d_out, nullptr, nullptr,
        H_DIM, 8, C_DIM, C_PAD, /*mode=*/1, /*reluOut=*/0);
}
